// RNNModel_84181359002003
// MI455X (gfx1250) — compile-verified, matrix operand order checked by reading, not run on hardware
//
#include <hip/hip_runtime.h>
#include <math.h>

typedef unsigned short u16;
typedef __attribute__((ext_vector_type(16))) __bf16 v16bf;
typedef __attribute__((ext_vector_type(8)))  float  v8f;
typedef __attribute__((ext_vector_type(2)))  float  v2f;

#define BN_SCALEF 0.9999950000375f   // 1/sqrt(1+1e-5)
#define NWG 16

// ---------------- helpers ----------------
__device__ __forceinline__ u16 f2bf(float f) {
  union { float f; unsigned u; } x; x.f = f;
  unsigned r = x.u + 0x7FFFu + ((x.u >> 16) & 1u);   // round-to-nearest-even
  return (u16)(r >> 16);
}

// Store one scalar into a bf16 buffer laid out as WMMA 16x16x32 bf16 A-tiles:
// buffer[kt][mt][lane][slot], lane: half*16+(m&15), slot encodes k per ISA table.
__device__ __forceinline__ void pack_bf(u16* base, int kbuf, int b, float val) {
  int kt = kbuf >> 5, kl = kbuf & 31;
  int hi = (kl >> 4) & 1, rem = kl & 15, hf = rem >> 3, rem2 = rem & 7;
  int slot = ((hi << 2) + (rem2 >> 1)) * 2 + (rem2 & 1);
  int lane = hf * 16 + (b & 15), mt = b >> 4;
  base[((((kt << 2) + mt) * 32 + lane) << 4) + slot] = f2bf(val);
}

// Store one scalar into an fp32 buffer laid out as WMMA 16x16x4 f32 A-tiles.
__device__ __forceinline__ void pack_f32(float* base, long k, int b, float val) {
  long kt = k >> 2; int kl = (int)(k & 3);
  int hf = kl >> 1, v = kl & 1;
  int lane = hf * 16 + (b & 15), mt = b >> 4;
  base[((kt * 4 + mt) * 32 + lane) * 2 + v] = val;
}

__device__ __forceinline__ float actfn(int type, float x) {
  if (type == 0) return tanhf(x);
  if (type == 1) return fmaxf(x, 0.f);
  if (type == 2) return 1.f / (1.f + __expf(-x));
  return x;
}

__device__ __forceinline__ void grid_sync(int* bar, int* s_sense) {
  __threadfence();
  __syncthreads();
  if (threadIdx.x == 0) {
    int sense = *s_sense ^ 1;
    *s_sense = sense;
    int old = __hip_atomic_fetch_add(&bar[0], 1, __ATOMIC_ACQ_REL, __HIP_MEMORY_SCOPE_AGENT);
    if (old == NWG - 1) {
      __hip_atomic_store(&bar[0], 0, __ATOMIC_RELAXED, __HIP_MEMORY_SCOPE_AGENT);
      __hip_atomic_store(&bar[1], sense, __ATOMIC_RELEASE, __HIP_MEMORY_SCOPE_AGENT);
    } else {
      while (__hip_atomic_load(&bar[1], __ATOMIC_ACQUIRE, __HIP_MEMORY_SCOPE_AGENT) != sense)
        __builtin_amdgcn_s_sleep(2);
    }
  }
  __syncthreads();
  __threadfence();
}

// ---------------- init ----------------
__global__ __launch_bounds__(256) void init_kernel(float* out_pre, int* bar) {
  for (int i = threadIdx.x; i < 64 * 700; i += 256) out_pre[i] = 0.f;
  if (threadIdx.x < 8) bar[threadIdx.x] = 0;
}

// ---------------- stem conv (k=13, pad=6) + BN affine ----------------
__global__ __launch_bounds__(256) void stem_kernel(const float* __restrict__ in,
    const float* __restrict__ w, const float* __restrict__ gamma,
    const float* __restrict__ beta, float* __restrict__ out) {
  __shared__ float ws_[52];
  __shared__ float sg, sb;
  int co = blockIdx.y, b = blockIdx.z;
  if (threadIdx.x < 52) ws_[threadIdx.x] = w[co * 52 + threadIdx.x];
  if (threadIdx.x == 0) { sg = gamma[co]; sb = beta[co]; }
  __syncthreads();
  int l = blockIdx.x * 256 + threadIdx.x;
  const float* inb = in + (size_t)b * 4 * 4096;
  float acc = 0.f;
  for (int ci = 0; ci < 4; ++ci)
    for (int k = 0; k < 13; ++k) {
      int p = l + k - 6;
      if (p >= 0 && p < 4096) acc += inb[ci * 4096 + p] * ws_[ci * 13 + k];
    }
  out[((size_t)b * 48 + co) * 4096 + l] = sg * (acc * BN_SCALEF) + sb;
}

// ---------------- cell0 (normal cell collapses to u * {1,2,4,8}) ----------------
__global__ __launch_bounds__(256) void cell0_kernel(const float* __restrict__ stem,
    const float* __restrict__ p0, const float* __restrict__ p1, float* __restrict__ out) {
  __shared__ float wsum[16 * 48];
  for (int i = threadIdx.x; i < 768; i += 256) wsum[i] = p0[i] + p1[i];
  __syncthreads();
  int b = blockIdx.y;
  int l = blockIdx.x * 256 + threadIdx.x;
  float acc[16];
#pragma unroll
  for (int j = 0; j < 16; ++j) acc[j] = 0.f;
  for (int i = 0; i < 48; ++i) {
    float xv = fmaxf(stem[((size_t)b * 48 + i) * 4096 + l], 0.f);
#pragma unroll
    for (int j = 0; j < 16; ++j) acc[j] += xv * wsum[j * 48 + i];
  }
  for (int j = 0; j < 16; ++j) {
    float u = acc[j] * BN_SCALEF;
    for (int g = 0; g < 4; ++g)
      out[((size_t)b * 64 + g * 16 + j) * 4096 + l] = u * (float)(1 << g);
  }
}

// ---------------- generic relu -> 1x1 conv -> BN ----------------
template <int CIN, int COUT>
__global__ __launch_bounds__(256) void rcb_kernel(const float* __restrict__ x,
    const float* __restrict__ w, float* __restrict__ out, int L) {
  __shared__ float ws_[CIN * COUT];
  for (int i = threadIdx.x; i < CIN * COUT; i += 256) ws_[i] = w[i];
  __syncthreads();
  int b = blockIdx.y;
  int l = blockIdx.x * 256 + threadIdx.x;
  float acc[COUT];
#pragma unroll
  for (int o = 0; o < COUT; ++o) acc[o] = 0.f;
  for (int i = 0; i < CIN; ++i) {
    float xv = fmaxf(x[((size_t)b * CIN + i) * L + l], 0.f);
#pragma unroll
    for (int o = 0; o < COUT; ++o) acc[o] += xv * ws_[o * CIN + i];
  }
  for (int o = 0; o < COUT; ++o)
    out[((size_t)b * COUT + o) * L + l] = acc[o] * BN_SCALEF;
}

// ---------------- standalone FactorizedReduce (cell2 preprocess0: 64ch,4096 -> 64ch,2048) ----
__global__ __launch_bounds__(256) void factred_kernel(const float* __restrict__ x,
    const float* __restrict__ w1, const float* __restrict__ w2, float* __restrict__ out) {
  __shared__ float ws_[2 * 32 * 64];
  for (int i = threadIdx.x; i < 2048; i += 256) { ws_[i] = w1[i]; ws_[2048 + i] = w2[i]; }
  __syncthreads();
  int b = blockIdx.y;
  int l = blockIdx.x * 256 + threadIdx.x;
  float a0[32], a1[32];
#pragma unroll
  for (int o = 0; o < 32; ++o) { a0[o] = 0.f; a1[o] = 0.f; }
  for (int i = 0; i < 64; ++i) {
    float xa = fmaxf(x[((size_t)b * 64 + i) * 4096 + 2 * l], 0.f);
    float xb = fmaxf(x[((size_t)b * 64 + i) * 4096 + 2 * l + 1], 0.f);
#pragma unroll
    for (int o = 0; o < 32; ++o) { a0[o] += xa * ws_[o * 64 + i]; a1[o] += xb * ws_[2048 + o * 64 + i]; }
  }
  for (int o = 0; o < 32; ++o) {
    out[((size_t)b * 64 + o) * 2048 + l]      = a0[o] * BN_SCALEF;
    out[((size_t)b * 64 + 32 + o) * 2048 + l] = a1[o] * BN_SCALEF;
  }
}

// ---------------- reduction cell: 8 FactorizedReduce edges + fixed integer combination ----
template <int CI, int COH>
__global__ __launch_bounds__(256) void reduce_cell_kernel(const float* __restrict__ pre0,
    const float* __restrict__ pre1, const float* __restrict__ fr,
    float* __restrict__ out, int Lout) {
  constexpr int TWO_CO = 2 * COH;
  constexpr int LPB = 256 / TWO_CO;
  __shared__ float sx[LPB][2][2][CI];
  int b = blockIdx.y;
  int l0 = blockIdx.x * LPB;
  int Lin = Lout * 2;
  for (int i = threadIdx.x; i < LPB * 4 * CI; i += 256) {
    int lp = i / (4 * CI); int rem = i % (4 * CI);
    int par = rem / (2 * CI); int rem2 = rem % (2 * CI);
    int which = rem2 / CI; int ci = rem2 % CI;
    const float* src = which ? pre1 : pre0;
    sx[lp][par][which][ci] = fmaxf(src[((size_t)b * CI + ci) * Lin + 2 * (l0 + lp) + par], 0.f);
  }
  __syncthreads();
  int c = threadIdx.x % TWO_CO;
  int lp = threadIdx.x / TWO_CO;
  int l = l0 + lp;
  int hf = c / COH, oo = c % COH;
  float F[8];
  for (int k = 0; k < 8; ++k) {
    const float* wv = fr + ((size_t)(k * 2 + hf) * COH + oo) * CI;
    const float* xr = sx[lp][hf][k & 1];
    float s = 0.f;
#pragma unroll 8
    for (int i = 0; i < CI; ++i) s += xr[i] * wv[i];
    F[k] = s * BN_SCALEF;
  }
  const int COEF[4][8] = {{1,1,0,0,0,0,0,0},{1,1,1,1,0,0,0,0},{2,2,1,1,1,1,0,0},{4,4,2,2,1,1,1,1}};
  for (int m = 0; m < 4; ++m) {
    float v = 0.f;
    for (int k = 0; k < 8; ++k) v += (float)COEF[m][k] * F[k];
    out[((size_t)b * (4 * TWO_CO) + m * TWO_CO + c) * Lout + l] = v;
  }
}

// ---------------- pre-pack x_t for RNN into bf16 WMMA A-tiles ----------------
__global__ __launch_bounds__(256) void packx_kernel(const float* __restrict__ cell2,
    u16* __restrict__ packedX) {
  int idx = blockIdx.x * 256 + threadIdx.x;           // 1024*64*256 elements
  int t = idx & 1023; int r = idx >> 10;
  int c = r & 255; int b = r >> 8;
  float val = cell2[(((size_t)b * 256 + c) << 10) + t];
  pack_bf(packedX + (size_t)t * 16384, c, b, val);
}

// ---------------- RNN scan: persistent 16-WG kernel, weights LDS-resident ----------------
__device__ __forceinline__ void rnn_mm(const u16* __restrict__ aLo, const u16* __restrict__ aHi,
    int KT, const u16* __restrict__ wNode, float* __restrict__ chLDS) {
  int lane = threadIdx.x & 31;
  int wave = threadIdx.x >> 5;
  int mt = wave & 3, nth = wave >> 2;
  v8f acc = {};
  for (int kt = 0; kt < KT; ++kt) {
    const u16* ap = (kt < 8) ? (aLo + ((((kt * 4 + mt) * 32) + lane) << 4))
                             : (aHi + (((((kt - 8) * 4 + mt) * 32) + lane) << 4));
    v16bf a = *(const v16bf*)ap;
    v16bf bm = *(const v16bf*)(wNode + ((nth * KT + kt) * 512) + (lane << 4));
    acc = __builtin_amdgcn_wmma_f32_16x16x32_bf16(false, a, false, bm, (short)0, acc, false, false);
  }
#pragma unroll
  for (int r = 0; r < 8; ++r) chLDS[((nth * 4 + mt) * 8 + r) * 32 + lane] = acc[r];
}

__device__ __forceinline__ void rnn_update(float* chLDS, float* ldsState, float* ldsSum,
    int predSlot, int dstSlot, int act, u16* packDst, int g, int sumMode) {
  for (int flat = threadIdx.x; flat < 1024; flat += 256) {
    int b = flat >> 4, j = flat & 15;
    int mt = b >> 4, rr = b & 15, r = rr & 7, hf = rr >> 3;
    int lane2 = hf * 16 + j;
    float cc = chLDS[(mt * 8 + r) * 32 + lane2];
    float hh = chLDS[((4 + mt) * 8 + r) * 32 + lane2];
    float sp = ldsState[predSlot * 1024 + flat];
    float sig = 1.f / (1.f + __expf(-cc));
    float s = sp + sig * (actfn(act, hh) - sp);
    if (dstSlot >= 0) ldsState[dstSlot * 1024 + flat] = s;
    if (sumMode == 2) ldsSum[flat] = 0.f;
    else if (sumMode == 1) ldsSum[flat] += s;
    if (packDst) pack_bf(packDst, g * 16 + j, b, s);
  }
}

__global__ __launch_bounds__(256) void rnn_kernel(const u16* __restrict__ packedX,
    const float* __restrict__ hidden, const float* __restrict__ W0,
    const float* __restrict__ Ws, u16* __restrict__ packedS, u16* __restrict__ packedH,
    float* __restrict__ decA, float* __restrict__ dout, int* __restrict__ bar) {
  extern __shared__ char smem[];
  u16*   ldsW     = (u16*)smem;               // 81920 bf16 (163840 B)
  float* chLDS    = (float*)(smem + 163840);  // 2048 f32
  float* ldsState = (float*)(smem + 172032);  // 6 x 1024 f32 (slot5 = h)
  float* ldsSum   = (float*)(smem + 196608);  // 1024 f32
  __shared__ int s_sense;

  const int g = blockIdx.x;
  const int tid = threadIdx.x;

  // Pack this WG's weight columns into LDS as WMMA B-tiles (bf16).
  for (int idx = tid; idx < 81920; idx += 256) {
    int tile = idx >> 9, e = idx & 511;
    int lane = e >> 4, slot = e & 15;
    int node, nt, kt;
    if (tile < 32) { node = 0; nt = tile >> 4; kt = tile & 15; }
    else { int r = tile - 32; node = 1 + (r >> 4); nt = (r >> 3) & 1; kt = r & 7; }
    int hf = lane >> 4;
    int v = slot >> 1, p = slot & 1;
    int kl = ((v & 3) << 1) + p + (hf << 3) + ((v >= 4) ? 16 : 0);
    int k = (kt << 5) + kl;
    int col = nt * 256 + g * 16 + (lane & 15);
    float wv = (node == 0) ? W0[k * 512 + col] : Ws[((node - 1) * 256 + k) * 512 + col];
    ldsW[idx] = f2bf(wv);
  }
  // init h from `hidden`
  for (int flat = tid; flat < 1024; flat += 256) {
    int b = flat >> 4, j = flat & 15;
    float h0 = hidden[b * 256 + g * 16 + j];
    ldsState[5 * 1024 + flat] = h0;
    pack_bf(packedH, g * 16 + j, b, h0);
  }
  if (tid == 0) s_sense = 0;
  grid_sync(bar, &s_sense);

  u16* pS0 = packedS;
  u16* pS1 = packedS + 16384;
  u16* pS2 = packedS + 32768;
  u16* pS3 = packedS + 49152;
  u16* pS5 = packedS + 65536;
#define WOFF(n) (ldsW + 16384 + ((n)-1) * 8192)

  for (int t = 0; t < 1024; ++t) {
    // Phase A: ch0 = [x_t, h] @ W0 ; s0 = h + sig(c)*(tanh(h') - h)
    rnn_mm(packedX + (size_t)t * 16384, packedH, 16, ldsW, chLDS);
    __syncthreads();
    rnn_update(chLDS, ldsState, ldsSum, 5, 0, 0, pS0, g, 2);
    grid_sync(bar, &s_sense);
    // L1: node0 (tanh, pred s0) -> s1
    rnn_mm(pS0, pS0, 8, WOFF(1), chLDS); __syncthreads();
    rnn_update(chLDS, ldsState, ldsSum, 0, 1, 0, pS1, g, 1);
    grid_sync(bar, &s_sense);
    // L2: node1(relu,s1)->s2, node2(relu,s1)->s3, node3(ident,s1)->s4
    rnn_mm(pS1, pS1, 8, WOFF(2), chLDS); __syncthreads();
    rnn_update(chLDS, ldsState, ldsSum, 1, 2, 1, pS2, g, 1); __syncthreads();
    rnn_mm(pS1, pS1, 8, WOFF(3), chLDS); __syncthreads();
    rnn_update(chLDS, ldsState, ldsSum, 1, 3, 1, pS3, g, 1); __syncthreads();
    rnn_mm(pS1, pS1, 8, WOFF(4), chLDS); __syncthreads();
    rnn_update(chLDS, ldsState, ldsSum, 1, -1, 3, (u16*)0, g, 1);
    grid_sync(bar, &s_sense);
    // L3: node4(tanh,s2)->s5, node6(tanh,s3)->s7
    rnn_mm(pS2, pS2, 8, WOFF(5), chLDS); __syncthreads();
    rnn_update(chLDS, ldsState, ldsSum, 2, 4, 0, pS5, g, 1); __syncthreads();
    rnn_mm(pS3, pS3, 8, WOFF(7), chLDS); __syncthreads();
    rnn_update(chLDS, ldsState, ldsSum, 3, -1, 0, (u16*)0, g, 1);
    grid_sync(bar, &s_sense);
    // L4: node5(sigmoid,s5)->s6, node7(relu,s5)->s8
    rnn_mm(pS5, pS5, 8, WOFF(6), chLDS); __syncthreads();
    rnn_update(chLDS, ldsState, ldsSum, 4, -1, 2, (u16*)0, g, 1); __syncthreads();
    rnn_mm(pS5, pS5, 8, WOFF(8), chLDS); __syncthreads();
    rnn_update(chLDS, ldsState, ldsSum, 4, -1, 1, (u16*)0, g, 1);
    __syncthreads();
    // finalize: h_new = mean(s1..s8); emit packedH + decoder-A (f32 WMMA layout)
    for (int flat = tid; flat < 1024; flat += 256) {
      int b = flat >> 4, j = flat & 15;
      float h = ldsSum[flat] * 0.125f;
      ldsState[5 * 1024 + flat] = h;
      pack_bf(packedH, g * 16 + j, b, h);
      pack_f32(decA, (long)t * 256 + g * 16 + j, b, h);
      if (t == 1023) dout[256 + b * 256 + g * 16 + j] = h;
    }
    grid_sync(bar, &s_sense);
  }
#undef WOFF
}

// ---------------- decoder GEMM: [64,262144] x dec_w^T (f32 WMMA 16x16x4) ----------------
__global__ __launch_bounds__(256) void dec_gemm_kernel(const float* __restrict__ A,
    const float* __restrict__ W, float* __restrict__ outp) {
  __shared__ float red[8 * 1024];
  int nt = blockIdx.x;
  int tid = threadIdx.x, lane = tid & 31, wave = tid >> 5;
  int hf = lane >> 4, nl = lane & 15;
  int n = nt * 16 + nl;
  float nmask = (n < 700) ? 1.f : 0.f;
  const float* wrow = W + (size_t)((n < 700) ? n : 0) * 262144;
  v8f c0 = {}, c1 = {}, c2 = {}, c3 = {};
  size_t ktS = (size_t)wave * 8192;
  for (size_t kt = ktS; kt < ktS + 8192; ++kt) {
    size_t kb = kt * 4 + (size_t)hf * 2;
    __builtin_prefetch(wrow + kb + 1024, 0, 1);
    v2f bv = *(const v2f*)(wrow + kb);
    bv *= nmask;
    const float* ab = A + kt * 256 + (size_t)lane * 2;
    v2f a0 = *(const v2f*)(ab);
    v2f a1 = *(const v2f*)(ab + 64);
    v2f a2 = *(const v2f*)(ab + 128);
    v2f a3 = *(const v2f*)(ab + 192);
    c0 = __builtin_amdgcn_wmma_f32_16x16x4_f32(false, a0, false, bv, (short)0, c0, false, false);
    c1 = __builtin_amdgcn_wmma_f32_16x16x4_f32(false, a1, false, bv, (short)0, c1, false, false);
    c2 = __builtin_amdgcn_wmma_f32_16x16x4_f32(false, a2, false, bv, (short)0, c2, false, false);
    c3 = __builtin_amdgcn_wmma_f32_16x16x4_f32(false, a3, false, bv, (short)0, c3, false, false);
  }
#pragma unroll
  for (int r = 0; r < 8; ++r) {
    red[wave * 1024 + (0 * 8 + r) * 32 + lane] = c0[r];
    red[wave * 1024 + (1 * 8 + r) * 32 + lane] = c1[r];
    red[wave * 1024 + (2 * 8 + r) * 32 + lane] = c2[r];
    red[wave * 1024 + (3 * 8 + r) * 32 + lane] = c3[r];
  }
  __syncthreads();
  for (int i = tid; i < 1024; i += 256) {    // deterministic fixed-order reduction
    float s = 0.f;
    for (int w8 = 0; w8 < 8; ++w8) s += red[w8 * 1024 + i];
    int mt = i >> 8, r = (i >> 5) & 7, ln = i & 31;
    int m = mt * 16 + r + 8 * (ln >> 4);
    int o = nt * 16 + (ln & 15);
    if (o < 700) outp[m * 700 + o] = s;
  }
}

// ---------------- head: relu(x@dec + b) @ cls^T + cls_b ----------------
__global__ __launch_bounds__(256) void head_kernel(const float* __restrict__ pre,
    const float* __restrict__ dec_b, const float* __restrict__ cls_w,
    const float* __restrict__ cls_b, float* __restrict__ dout) {
  int b = threadIdx.x >> 2, j = threadIdx.x & 3;
  float s = cls_b[j];
  for (int o = 0; o < 700; ++o) {
    float x = fmaxf(pre[b * 700 + o] + dec_b[o], 0.f);
    s += x * cls_w[j * 700 + o];
  }
  dout[b * 4 + j] = s;
}

// ---------------- launch ----------------
extern "C" void kernel_launch(void* const* d_in, const int* in_sizes, int n_in,
                              void* d_out, int out_size, void* d_ws, size_t ws_size,
                              hipStream_t stream) {
  const float* in     = (const float*)d_in[0];
  const float* hidden = (const float*)d_in[1];
  const float* stem_w = (const float*)d_in[2];
  const float* stem_g = (const float*)d_in[3];
  const float* stem_b = (const float*)d_in[4];
  const float* c0p0   = (const float*)d_in[5];
  const float* c0p1   = (const float*)d_in[6];
  const float* c1p0   = (const float*)d_in[7];
  const float* c1p1   = (const float*)d_in[8];
  const float* c1fr   = (const float*)d_in[9];
  const float* c2p0a  = (const float*)d_in[10];
  const float* c2p0b  = (const float*)d_in[11];
  const float* c2p1   = (const float*)d_in[12];
  const float* c2fr   = (const float*)d_in[13];
  const float* W0     = (const float*)d_in[14];
  const float* Ws     = (const float*)d_in[15];
  const float* dec_w  = (const float*)d_in[16];
  const float* dec_b  = (const float*)d_in[17];
  const float* cls_w  = (const float*)d_in[18];
  const float* cls_b  = (const float*)d_in[19];
  (void)in_sizes; (void)n_in; (void)out_size; (void)ws_size;

  char* ws = (char*)d_ws;
  float* S1 = (float*)(ws + 0);           // 67 MB slots, aliased over the pipeline
  float* S2 = (float*)(ws + 67108864);
  float* S3 = (float*)(ws + 134217728);
  float* S4 = (float*)(ws + 201326592);
  float* S5 = (float*)(ws + 268435456);
  float* S6 = (float*)(ws + 301989888);
  char* tail = ws + 335544320;
  u16*   packedS = (u16*)tail;            // 5 x 16384 bf16
  u16*   packedH = (u16*)(tail + 163840); // 16384 bf16
  float* out_pre = (float*)(tail + 196608);
  int*   bar     = (int*)(tail + 376064);
  float* outF = (float*)d_out;

  init_kernel<<<1, 256, 0, stream>>>(out_pre, bar);
  stem_kernel<<<dim3(16, 48, 64), 256, 0, stream>>>(in, stem_w, stem_g, stem_b, S1);
  cell0_kernel<<<dim3(16, 64), 256, 0, stream>>>(S1, c0p0, c0p1, S2);
  rcb_kernel<48, 32><<<dim3(16, 64), 256, 0, stream>>>(S1, c1p0, S5, 4096);
  rcb_kernel<64, 32><<<dim3(16, 64), 256, 0, stream>>>(S2, c1p1, S6, 4096);
  reduce_cell_kernel<32, 16><<<dim3(256, 64), 256, 0, stream>>>(S5, S6, c1fr, S3, 2048);
  factred_kernel<<<dim3(8, 64), 256, 0, stream>>>(S2, c2p0a, c2p0b, S1);
  rcb_kernel<128, 64><<<dim3(8, 64), 256, 0, stream>>>(S3, c2p1, S5, 2048);
  reduce_cell_kernel<64, 32><<<dim3(256, 64), 256, 0, stream>>>(S1, S5, c2fr, S4, 1024);

  u16* packedX = (u16*)S2;                // cell0 buffer dead -> reuse
  packx_kernel<<<65536, 256, 0, stream>>>(S4, packedX);

  float* decA = S3;                       // cell1 buffer dead -> reuse (exactly 64 MB)
  size_t smem = 200704;                   // 160KB weights + staging (WGP has 320KB)
  hipFuncSetAttribute(reinterpret_cast<const void*>(rnn_kernel),
                      hipFuncAttributeMaxDynamicSharedMemorySize, (int)smem);
  rnn_kernel<<<NWG, 256, smem, stream>>>(packedX, hidden, W0, Ws, packedS, packedH,
                                         decA, outF, bar);
  dec_gemm_kernel<<<44, 256, 0, stream>>>(decA, dec_w, out_pre);
  head_kernel<<<1, 256, 0, stream>>>(out_pre, dec_b, cls_w, cls_b, outF);
}
